// Attention_21998822490697
// MI455X (gfx1250) — compile-verified, matrix operand order checked by reading, not run on hardware
//
#include <hip/hip_runtime.h>
#include <hip/hip_bf16.h>
#include <math.h>

// ---------------------------------------------------------------------------
// MI455X (gfx1250) multi-head attention, bf16 WMMA pipeline.
//   B=2, N=2048, E=1024, H=16, HD=64.  ~69 GFLOP vs ~50MB data -> compute
// bound; all matmuls via v_wmma_f32_16x16x32_bf16.  Tile staging uses the
// CDNA5 async copy path (global_load_async_to_lds_b128 + s_wait_asynccnt)
// with double-buffered LDS so DMA overlaps WMMA.
// ---------------------------------------------------------------------------

#define BATCH 2
#define SEQ   2048
#define EMB   1024
#define HEADS 16
#define HD    64
#define M_ALL (BATCH * SEQ)        // 4096
#define QKV_N (3 * EMB)            // 3072

typedef __attribute__((ext_vector_type(16))) __bf16 v16bf;
typedef __attribute__((ext_vector_type(8)))  float  v8f;

union Frag {
    v16bf v;
    uint4 q[2];
};

__device__ __forceinline__ unsigned short f2bf(float f) {
    unsigned u = __float_as_uint(f);
    unsigned r = u + 0x7FFFu + ((u >> 16) & 1u);   // round-to-nearest-even
    return (unsigned short)(r >> 16);
}

// Async DMA of 16 bytes global -> LDS (no VGPR round-trip, ASYNCcnt-tracked).
// Low 32 bits of a generic pointer to LDS are the wave-relative LDS byte
// address.
__device__ __forceinline__ void async_copy16(unsigned short* ldsp,
                                             const unsigned short* gp) {
    unsigned laddr = (unsigned)(size_t)(ldsp);
    asm volatile("global_load_async_to_lds_b128 %0, %1, off"
                 :: "v"(laddr), "v"(gp)
                 : "memory");
}
__device__ __forceinline__ void wait_async0() {
    asm volatile("s_wait_asynccnt 0" ::: "memory");
}

// ---------------------------------------------------------------------------
// Kernel 1: fp32 -> bf16 pack (row-major, k-contiguous)
// ---------------------------------------------------------------------------
__global__ void pack_bf16_kernel(const float* __restrict__ in,
                                 unsigned short* __restrict__ out, int n) {
    int i = blockIdx.x * blockDim.x + threadIdx.x;
    if (i < n) out[i] = f2bf(in[i]);
}

// ---------------------------------------------------------------------------
// Kernel 2: W [K][Ncols] fp32 -> WT [Ncols][K] bf16  (LDS 32x32 tile transpose)
// ---------------------------------------------------------------------------
__global__ void transpose_pack_kernel(const float* __restrict__ W,
                                      unsigned short* __restrict__ WT,
                                      int K, int Ncols) {
    __shared__ float tile[32][33];
    int bx = blockIdx.x, by = blockIdx.y;
    int tx = threadIdx.x, ty = threadIdx.y;
    for (int i = 0; i < 32; i += 8)
        tile[ty + i][tx] = W[(size_t)(by * 32 + ty + i) * Ncols + bx * 32 + tx];
    __syncthreads();
    for (int i = 0; i < 32; i += 8)
        WT[(size_t)(bx * 32 + ty + i) * K + by * 32 + tx] = f2bf(tile[tx][ty + i]);
}

// ---------------------------------------------------------------------------
// GEMM core: C(128x128) = A(128xK) * BT(128 rows x K)^T.
// LDS double-buffered (2 x 16KB); async DMA of tile k+1 overlaps WMMA of k.
// 8 waves, each 32x64 = 2x4 grid of 16x16x32 bf16 WMMAs.
// Fragment per lane: row = lane%16, two 16B k-runs at half*8 and 16+half*8.
// ---------------------------------------------------------------------------
__device__ __forceinline__ void gemm_stage(const unsigned short* __restrict__ A,
                                           const unsigned short* __restrict__ BT,
                                           int m0, int n0, int K, int k0,
                                           unsigned short* buf, int tid) {
#pragma unroll
    for (int s = 0; s < 2; ++s) {
        int i = tid + s * 256;            // 0..511
        int row = i >> 2, off = i & 3;
        async_copy16(buf + row * 32 + off * 8,
                     A + (size_t)(m0 + row) * K + k0 + off * 8);
    }
#pragma unroll
    for (int s = 0; s < 2; ++s) {
        int i = tid + s * 256;
        int row = i >> 2, off = i & 3;
        async_copy16(buf + 4096 + row * 32 + off * 8,
                     BT + (size_t)(n0 + row) * K + k0 + off * 8);
    }
}

__device__ __forceinline__ void gemm_core(const unsigned short* __restrict__ A,
                                          const unsigned short* __restrict__ BT,
                                          int m0, int n0, int K,
                                          unsigned short* lds,   // 2*8192 u16
                                          v8f acc[2][4]) {
    const int tid  = threadIdx.x;
    const int wave = tid >> 5, lane = tid & 31;
    const int wm   = wave >> 1, wn = wave & 1;
    const int half = lane >> 4, l16 = lane & 15;

    for (int i = 0; i < 2; ++i)
        for (int j = 0; j < 4; ++j)
            for (int r = 0; r < 8; ++r) acc[i][j][r] = 0.0f;

    gemm_stage(A, BT, m0, n0, K, 0, lds, tid);
    wait_async0();
    __syncthreads();

    for (int k0 = 0; k0 < K; k0 += 32) {
        unsigned short* cur = lds + ((k0 >> 5) & 1) * 8192;
        unsigned short* nxt = lds + (((k0 >> 5) + 1) & 1) * 8192;
        if (k0 + 32 < K)
            gemm_stage(A, BT, m0, n0, K, k0 + 32, nxt, tid);   // async, overlaps

        Frag a[2];
#pragma unroll
        for (int i = 0; i < 2; ++i) {
            int base = (wm * 32 + i * 16 + l16) * 32;
            a[i].q[0] = *(const uint4*)(cur + base + half * 8);
            a[i].q[1] = *(const uint4*)(cur + base + 16 + half * 8);
        }
#pragma unroll
        for (int j = 0; j < 4; ++j) {
            int base = 4096 + (wn * 64 + j * 16 + l16) * 32;
            Frag b;
            b.q[0] = *(const uint4*)(cur + base + half * 8);
            b.q[1] = *(const uint4*)(cur + base + 16 + half * 8);
#pragma unroll
            for (int i = 0; i < 2; ++i)
                acc[i][j] = __builtin_amdgcn_wmma_f32_16x16x32_bf16(
                    false, a[i].v, false, b.v, (short)0, acc[i][j], false, false);
        }
        wait_async0();
        __syncthreads();
    }
}

// ---------------------------------------------------------------------------
// Kernel 3: QKV GEMM, scatter into Q[bh][n][d], K[bh][n][d], VT[bh][d][n] bf16
// ---------------------------------------------------------------------------
__global__ void qkv_gemm_kernel(const unsigned short* __restrict__ XB,
                                const unsigned short* __restrict__ WT,
                                const float* __restrict__ bias,
                                unsigned short* __restrict__ Qo,
                                unsigned short* __restrict__ Ko,
                                unsigned short* __restrict__ Vt) {
    __shared__ __align__(16) unsigned short lds[2 * 8192];
    v8f acc[2][4];
    const int m0 = blockIdx.y * 128, n0 = blockIdx.x * 128;
    gemm_core(XB, WT, m0, n0, EMB, lds, acc);

    const int tid  = threadIdx.x;
    const int wave = tid >> 5, lane = tid & 31;
    const int wm   = wave >> 1, wn = wave & 1;
    const int half = lane >> 4, l16 = lane & 15;

    for (int i = 0; i < 2; ++i)
        for (int j = 0; j < 4; ++j) {
            int c = n0 + wn * 64 + j * 16 + l16;
            int hh = c / 192, rem = c % 192, part = rem / 64, d = rem % 64;
            float bv = bias[c];
            for (int r = 0; r < 8; ++r) {
                int m = m0 + wm * 32 + i * 16 + r + half * 8;
                int bb = m / SEQ, n = m % SEQ;
                unsigned short bf = f2bf(acc[i][j][r] + bv);
                size_t bh = (size_t)(bb * HEADS + hh);
                if (part == 0)      Qo[(bh * SEQ + n) * HD + d] = bf;
                else if (part == 1) Ko[(bh * SEQ + n) * HD + d] = bf;
                else                Vt[(bh * HD + d) * SEQ + n] = bf;
            }
        }
}

// ---------------------------------------------------------------------------
// Kernel 4: flash-style attention. Block = (128 q-rows, one bh); each wave
// owns 16 q-rows. K/V 64-key chunks are async-DMA'd into double-buffered LDS
// (shared by all 8 waves); P spilled bf16 to wave-private LDS for the P*V
// WMMAs; online softmax in fp32 with wave32 shuffles.
// ---------------------------------------------------------------------------
__global__ void attn_kernel(const unsigned short* __restrict__ Qb_,
                            const unsigned short* __restrict__ Kb_,
                            const unsigned short* __restrict__ Vt_,
                            const float* __restrict__ mask,
                            unsigned short* __restrict__ AO) {
    __shared__ __align__(16) unsigned short kvlds[2 * 8192]; // [buf][K:4096|V:4096]
    __shared__ __align__(16) unsigned short plds[8 * 1024];  // 8 waves * 16x64 bf16
    const int qb = blockIdx.x, bh = blockIdx.y;
    const int b = bh >> 4, hh = bh & 15;
    const int tid  = threadIdx.x;
    const int wave = tid >> 5, lane = tid & 31;
    const int half = lane >> 4, l16 = lane & 15;
    unsigned short* wp = plds + wave * 1024;

    const unsigned short* Qp = Qb_ + (size_t)bh * SEQ * HD;
    const unsigned short* Kp = Kb_ + (size_t)bh * SEQ * HD;
    const unsigned short* Vp = Vt_ + (size_t)bh * HD * SEQ;

    // Q fragments for this wave's 16 rows (resident in VGPRs for whole kernel)
    const int qrow = qb * 128 + wave * 16 + l16;
    Frag qf[2];
#pragma unroll
    for (int kk = 0; kk < 2; ++kk) {
        const unsigned short* p = Qp + (size_t)qrow * HD + kk * 32 + half * 8;
        qf[kk].q[0] = *(const uint4*)(p);
        qf[kk].q[1] = *(const uint4*)(p + 16);
    }

    // stage K/V chunk [j0, j0+64) into buf: K rows contiguous (8KB block),
    // V^T rows are 64 u16 slices of SEQ-long rows.
    auto stage_kv = [&](unsigned short* buf, int j0) {
        const unsigned short* kg = Kp + (size_t)j0 * HD;
#pragma unroll
        for (int s = 0; s < 2; ++s) {
            int i = tid + s * 256;             // 0..511, 16B chunks
            async_copy16(buf + i * 8, kg + i * 8);
        }
#pragma unroll
        for (int s = 0; s < 2; ++s) {
            int i = tid + s * 256;
            int d = i >> 3, off = i & 7;
            async_copy16(buf + 4096 + d * 64 + off * 8,
                         Vp + (size_t)d * SEQ + j0 + off * 8);
        }
    };

    float rm[8], lsum[8];
    v8f o[4];
    for (int r = 0; r < 8; ++r) { rm[r] = -INFINITY; lsum[r] = 0.0f; }
    for (int t = 0; t < 4; ++t)
        for (int r = 0; r < 8; ++r) o[t][r] = 0.0f;

    stage_kv(kvlds, 0);
    wait_async0();
    __syncthreads();

    for (int j0 = 0; j0 < SEQ; j0 += 64) {
        unsigned short* cur = kvlds + ((j0 >> 6) & 1) * 8192;
        unsigned short* nxt = kvlds + (((j0 >> 6) + 1) & 1) * 8192;
        if (j0 + 64 < SEQ)
            stage_kv(nxt, j0 + 64);            // async, overlaps compute

        // ---- S = Q * K^T (16x64 per wave), K frags from LDS ----
        v8f s[4];
        for (int t = 0; t < 4; ++t)
            for (int r = 0; r < 8; ++r) s[t][r] = 0.0f;
#pragma unroll
        for (int kk = 0; kk < 2; ++kk)
#pragma unroll
            for (int t = 0; t < 4; ++t) {
                int base = (t * 16 + l16) * 64 + kk * 32 + half * 8;
                Frag kf;
                kf.q[0] = *(const uint4*)(cur + base);
                kf.q[1] = *(const uint4*)(cur + base + 16);
                s[t] = __builtin_amdgcn_wmma_f32_16x16x32_bf16(
                    false, qf[kk].v, false, kf.v, (short)0, s[t], false, false);
            }
        // ---- + attention_mask ----
        for (int t = 0; t < 4; ++t)
            for (int r = 0; r < 8; ++r) {
                int nrow = qb * 128 + wave * 16 + r + half * 8;
                s[t][r] += mask[(size_t)nrow * SEQ + j0 + t * 16 + l16];
            }
        // ---- online softmax ----
        float mx[8];
        for (int r = 0; r < 8; ++r) {
            float v = s[0][r];
            for (int t = 1; t < 4; ++t) v = fmaxf(v, s[t][r]);
            for (int msk = 1; msk < 16; msk <<= 1)
                v = fmaxf(v, __shfl_xor(v, msk, 32));
            mx[r] = v;
        }
        float scale[8], rs[8];
        for (int r = 0; r < 8; ++r) {
            float nm = fmaxf(rm[r], mx[r]);
            scale[r] = (rm[r] == -INFINITY) ? 0.0f : __expf(rm[r] - nm);
            rm[r] = nm;
            rs[r] = 0.0f;
        }
        for (int t = 0; t < 4; ++t)
            for (int r = 0; r < 8; ++r) {
                float pv = __expf(s[t][r] - rm[r]);
                rs[r] += pv;
                wp[(r + half * 8) * 64 + t * 16 + l16] = f2bf(pv);
            }
        for (int r = 0; r < 8; ++r) {
            float v = rs[r];
            for (int msk = 1; msk < 16; msk <<= 1) v += __shfl_xor(v, msk, 32);
            lsum[r] = lsum[r] * scale[r] + v;
        }
        for (int t = 0; t < 4; ++t)
            for (int r = 0; r < 8; ++r) o[t][r] *= scale[r];
        asm volatile("s_wait_dscnt 0" ::: "memory");   // P spill RAW (same wave)

        // ---- O = O*scale + P * V, V frags from LDS ----
#pragma unroll
        for (int kk = 0; kk < 2; ++kk) {
            Frag pf;
            const unsigned short* pp = wp + l16 * 64 + kk * 32 + half * 8;
            pf.q[0] = *(const uint4*)(pp);
            pf.q[1] = *(const uint4*)(pp + 16);
#pragma unroll
            for (int t = 0; t < 4; ++t) {
                int base = 4096 + (t * 16 + l16) * 64 + kk * 32 + half * 8;
                Frag vf;
                vf.q[0] = *(const uint4*)(cur + base);
                vf.q[1] = *(const uint4*)(cur + base + 16);
                o[t] = __builtin_amdgcn_wmma_f32_16x16x32_bf16(
                    false, pf.v, false, vf.v, (short)0, o[t], false, false);
            }
        }
        wait_async0();
        __syncthreads();
    }
    // ---- normalize and write attn output bf16 [b][n][E] ----
    for (int t = 0; t < 4; ++t)
        for (int r = 0; r < 8; ++r) {
            int nrow = qb * 128 + wave * 16 + r + half * 8;
            float val = o[t][r] / lsum[r];
            AO[((size_t)(b * SEQ + nrow)) * EMB + hh * HD + t * 16 + l16] = f2bf(val);
        }
}

// ---------------------------------------------------------------------------
// Kernel 5: output projection -> fp32 d_out [B][N][E]
// ---------------------------------------------------------------------------
__global__ void proj_gemm_kernel(const unsigned short* __restrict__ AO,
                                 const unsigned short* __restrict__ WT,
                                 const float* __restrict__ bias,
                                 float* __restrict__ out) {
    __shared__ __align__(16) unsigned short lds[2 * 8192];
    v8f acc[2][4];
    const int m0 = blockIdx.y * 128, n0 = blockIdx.x * 128;
    gemm_core(AO, WT, m0, n0, EMB, lds, acc);

    const int tid  = threadIdx.x;
    const int wave = tid >> 5, lane = tid & 31;
    const int wm   = wave >> 1, wn = wave & 1;
    const int half = lane >> 4, l16 = lane & 15;

    for (int i = 0; i < 2; ++i)
        for (int j = 0; j < 4; ++j) {
            int c = n0 + wn * 64 + j * 16 + l16;
            float bv = bias[c];
            for (int r = 0; r < 8; ++r) {
                int m = m0 + wm * 32 + i * 16 + r + half * 8;
                out[(size_t)m * EMB + c] = acc[i][j][r] + bv;
            }
        }
}

// ---------------------------------------------------------------------------
// Host-side launch
// ---------------------------------------------------------------------------
extern "C" void kernel_launch(void* const* d_in, const int* in_sizes, int n_in,
                              void* d_out, int out_size, void* d_ws, size_t ws_size,
                              hipStream_t stream) {
    const float* x     = (const float*)d_in[0];
    const float* mask  = (const float*)d_in[1];
    const float* Wqkv  = (const float*)d_in[2];
    const float* bqkv  = (const float*)d_in[3];
    const float* Wproj = (const float*)d_in[4];
    const float* bproj = (const float*)d_in[5];
    float* out = (float*)d_out;

    unsigned short* ws = (unsigned short*)d_ws;
    unsigned short* XB     = ws;                                  // 4096x1024
    unsigned short* WQKVT  = XB     + (size_t)M_ALL * EMB;        // 3072x1024
    unsigned short* WPROJT = WQKVT  + (size_t)QKV_N * EMB;        // 1024x1024
    unsigned short* Qd     = WPROJT + (size_t)EMB * EMB;          // 32x2048x64
    unsigned short* Kd     = Qd     + (size_t)BATCH * HEADS * SEQ * HD;
    unsigned short* VTd    = Kd     + (size_t)BATCH * HEADS * SEQ * HD;
    unsigned short* AOd    = VTd    + (size_t)BATCH * HEADS * SEQ * HD;
    // total: 48 MB of workspace

    {
        int n = M_ALL * EMB;
        pack_bf16_kernel<<<(n + 255) / 256, 256, 0, stream>>>(x, XB, n);
    }
    {
        dim3 blk(32, 8);
        transpose_pack_kernel<<<dim3(QKV_N / 32, EMB / 32), blk, 0, stream>>>(Wqkv, WQKVT, EMB, QKV_N);
        transpose_pack_kernel<<<dim3(EMB / 32, EMB / 32), blk, 0, stream>>>(Wproj, WPROJT, EMB, EMB);
    }
    qkv_gemm_kernel<<<dim3(QKV_N / 128, M_ALL / 128), 256, 0, stream>>>(
        XB, WQKVT, bqkv, Qd, Kd, VTd);
    attn_kernel<<<dim3(SEQ / 128, BATCH * HEADS), 256, 0, stream>>>(
        Qd, Kd, VTd, mask, AOd);
    proj_gemm_kernel<<<dim3(EMB / 128, M_ALL / 128), 256, 0, stream>>>(
        AOd, WPROJT, bproj, out);
}
